// EmbeddingModule_73735998538459
// MI455X (gfx1250) — matrix operand order repaired and checked by reading, not compiled, not run on hardware
//
#include <hip/hip_runtime.h>
#include <hip/hip_bf16.h>

typedef __attribute__((ext_vector_type(16))) _Float16 v16h;
typedef __attribute__((ext_vector_type(8)))  _Float16 v8h;
typedef __attribute__((ext_vector_type(8)))  float    v8f;
typedef __attribute__((ext_vector_type(4)))  unsigned int u32x4;
typedef __attribute__((ext_vector_type(8)))  unsigned int u32x8;
typedef __attribute__((address_space(3)))    _Float16 lds_f16_t;

#define B_      8
#define N_      2048
#define FD_     256
#define NPOINT_ 1024
#define NSAMPLE_ 16
#define RADIUS_ 0.3f
#define CIN_    259     // 3 + 256
#define KT1     9       // 8 feature k-tiles (TDM gather) + 1 xyz/pad k-tile
#define KT2     4
#define NGROUP_ (B_ * NPOINT_)

// ---------------------------------------------------------------------------
// Kernel 1: farthest point sampling, one workgroup per batch (sequential dep
// chain -> keep xyz and running min-distance resident in LDS).
// ---------------------------------------------------------------------------
__launch_bounds__(256)
__global__ void fps_kernel(const float* __restrict__ xyz,
                           int* __restrict__ inds, float* __restrict__ nxyz) {
    __shared__ float xs[N_], ys[N_], zs[N_], dist[N_];
    __shared__ float rv[256];
    __shared__ int   ri[256];
    __shared__ int   s_last;
    const int b = blockIdx.x;
    const int t = threadIdx.x;
    const float* xb = xyz + (size_t)b * N_ * 3;
    for (int n = t; n < N_; n += 256) {
        xs[n] = xb[n * 3 + 0];
        ys[n] = xb[n * 3 + 1];
        zs[n] = xb[n * 3 + 2];
        dist[n] = 1e10f;
    }
    if (t == 0) s_last = 0;
    __syncthreads();
    for (int i = 0; i < NPOINT_; ++i) {
        const int last = s_last;
        if (t == 0) {
            inds[b * NPOINT_ + i] = last;
            nxyz[(b * NPOINT_ + i) * 3 + 0] = xs[last];
            nxyz[(b * NPOINT_ + i) * 3 + 1] = ys[last];
            nxyz[(b * NPOINT_ + i) * 3 + 2] = zs[last];
        }
        const float px = xs[last], py = ys[last], pz = zs[last];
        float best = -1.0f; int bestIdx = 0;
        for (int n = t; n < N_; n += 256) {
            float dx = xs[n] - px, dy = ys[n] - py, dz = zs[n] - pz;
            float d = dx * dx + dy * dy + dz * dz;
            float nd = fminf(dist[n], d);
            dist[n] = nd;
            if (nd > best || (nd == best && n < bestIdx)) { best = nd; bestIdx = n; }
        }
        rv[t] = best; ri[t] = bestIdx;
        __syncthreads();
        for (int s = 128; s > 0; s >>= 1) {
            if (t < s) {
                float ov = rv[t + s]; int oi = ri[t + s];
                if (ov > rv[t] || (ov == rv[t] && oi < ri[t])) { rv[t] = ov; ri[t] = oi; }
            }
            __syncthreads();
        }
        if (t == 0) s_last = ri[0];
        __syncthreads();
    }
}

// ---------------------------------------------------------------------------
// Kernel 2a: transpose features (B, 256, N) f32 -> featT (B, N, 256) f16 so
// each neighbor's channels are one contiguous 512B row (TDM-gatherable).
// ---------------------------------------------------------------------------
__global__ void prep_feat_kernel(const float* __restrict__ feat,
                                 _Float16* __restrict__ featT) {
    size_t tid = (size_t)blockIdx.x * blockDim.x + threadIdx.x;
    if (tid >= (size_t)B_ * N_ * FD_) return;
    int c = (int)(tid & (FD_ - 1));
    size_t bn = tid >> 8;
    int n = (int)(bn & (N_ - 1));
    int b = (int)(bn >> 11);
    featT[tid] = (_Float16)feat[((size_t)b * FD_ + c) * N_ + n];
}

// ---------------------------------------------------------------------------
// Kernel 2b: pre-swizzle W1/W2/W3 into WMMA B-operand fragment layout (f16):
// frag[(nt*KT+kt)*32+lane][e] corresponds to output n = nt*16 + lane%16 and
// K' = kt*32 + (lane/16)*16 + e, where layer-1 uses permuted channels:
//   K' in [0,256)  -> original channel 3+K' (features, matches TDM tile)
//   K' in [256,259)-> original channel K'-256 (relative xyz)
//   K' >= 259      -> zero pad
// ---------------------------------------------------------------------------
__global__ void prep_weights_kernel(const float* __restrict__ W1,
                                    const float* __restrict__ W2,
                                    const float* __restrict__ W3,
                                    _Float16* __restrict__ w1f,
                                    _Float16* __restrict__ w2f,
                                    _Float16* __restrict__ w3f) {
    const int SZ1 = 8 * KT1 * 32 * 16;   // 36864
    const int SZ2 = 8 * KT2 * 32 * 16;   // 16384
    int tid = blockIdx.x * blockDim.x + threadIdx.x;
    if (tid < SZ1) {
        int e = tid & 15, lane = (tid >> 4) & 31, ktnt = tid >> 9;
        int kt = ktnt % KT1, nt = ktnt / KT1;
        int n = nt * 16 + (lane & 15);
        int Kp = kt * 32 + (lane >> 4) * 16 + e;
        int c = (Kp < 256) ? (3 + Kp) : (Kp - 256);
        float v = (Kp < CIN_) ? W1[n * CIN_ + c] : 0.0f;
        w1f[tid] = (_Float16)v;
    } else if (tid < SZ1 + SZ2) {
        int r = tid - SZ1;
        int e = r & 15, lane = (r >> 4) & 31, ktnt = r >> 9;
        int kt = ktnt % KT2, nt = ktnt / KT2;
        int n = nt * 16 + (lane & 15);
        int K = kt * 32 + (lane >> 4) * 16 + e;
        w2f[r] = (_Float16)W2[n * 128 + K];
    } else if (tid < SZ1 + 2 * SZ2) {
        int r = tid - SZ1 - SZ2;
        int e = r & 15, lane = (r >> 4) & 31, ktnt = r >> 9;
        int kt = ktnt % KT2, nt = ktnt / KT2;
        int n = nt * 16 + (lane & 15);
        int K = kt * 32 + (lane >> 4) * 16 + e;
        w3f[r] = (_Float16)W3[n * 128 + K];
    }
}

// ---------------------------------------------------------------------------
// Kernel 3: ball query — one wave per sampled point; collect first NSAMPLE
// in-radius indices in ascending order via ballot + prefix popcount.
// ---------------------------------------------------------------------------
__launch_bounds__(256)
__global__ void ballquery_kernel(const float* __restrict__ xyz,
                                 const float* __restrict__ nxyz,
                                 int* __restrict__ idx) {
    const int wave = (blockIdx.x * blockDim.x + threadIdx.x) >> 5;
    const int lane = threadIdx.x & 31;
    const int b = wave / NPOINT_;
    const float* xb = xyz + (size_t)b * N_ * 3;
    const float cx = nxyz[wave * 3 + 0];
    const float cy = nxyz[wave * 3 + 1];
    const float cz = nxyz[wave * 3 + 2];
    int* out = idx + wave * NSAMPLE_;
    int cnt = 0;
    int firstIdx = -1;
    const float r2 = RADIUS_ * RADIUS_;
    for (int base = 0; base < N_ && cnt < NSAMPLE_; base += 32) {
        int n = base + lane;
        float dx = xb[n * 3 + 0] - cx;
        float dy = xb[n * 3 + 1] - cy;
        float dz = xb[n * 3 + 2] - cz;
        bool valid = (dx * dx + dy * dy + dz * dz) < r2;
        unsigned int bal = (unsigned int)__ballot(valid);
        if (firstIdx < 0 && bal) firstIdx = base + __ffs(bal) - 1;
        unsigned int lt = (1u << lane) - 1u;
        int slot = cnt + __popc(bal & lt);
        if (valid && slot < NSAMPLE_) out[slot] = n;
        cnt += __popc(bal);
    }
    if (cnt > NSAMPLE_) cnt = NSAMPLE_;
    if (firstIdx < 0) firstIdx = 0;           // cannot happen (center is valid)
    if (lane < NSAMPLE_ && lane >= cnt) out[lane] = firstIdx;
}

// ---------------------------------------------------------------------------
// Kernel 4: grouped MLP via WMMA, activations staged by the Tensor Data Mover
// in gather mode: one tensor_load_to_lds per group pulls 16 indexed 512B rows
// of f16 features straight into LDS (TENSORcnt-tracked), while the xyz/pad
// k-tile (32 channels) is built by the VALU. 8 waves = 8 output n-tiles.
// ---------------------------------------------------------------------------
__launch_bounds__(256)
__global__ void mlp_kernel(const float* __restrict__ xyz,
                           const _Float16* __restrict__ featT,
                           const float* __restrict__ nxyz,
                           const int* __restrict__ idx,
                           const _Float16* __restrict__ w1f,
                           const _Float16* __restrict__ w2f,
                           const _Float16* __restrict__ w3f,
                           const float* __restrict__ b1,
                           const float* __restrict__ b2,
                           const float* __restrict__ b3,
                           float* __restrict__ agg) {
    __shared__ __align__(16) _Float16 h0feat[16][256];   // filled by TDM gather
    __shared__ __align__(16) _Float16 h0x[16][32];       // xyz-rel + zero pad
    __shared__ __align__(16) _Float16 h1[16][128];
    __shared__ __align__(16) _Float16 h2[16][128];
    __shared__ int sidx[16];

    const int g = blockIdx.x;             // group id, 0..8191
    const int b = g >> 10;
    const int t = threadIdx.x;
    const int lane = t & 31;
    const int nt = t >> 5;                // wave id == output n-tile
    const int half = lane >> 4;
    const int ncol = lane & 15;

    if (t < NSAMPLE_) sidx[t] = idx[g * NSAMPLE_ + t];
    __syncthreads();

    // xyz / pad k-tile: 16 rows x 32 cols
    const float cx = nxyz[g * 3 + 0];
    const float cy = nxyz[g * 3 + 1];
    const float cz = nxyz[g * 3 + 2];
    for (int e = t; e < 16 * 32; e += 256) {
        int j = e >> 5, c = e & 31;
        float v = 0.0f;
        if (c < 3) {
            const float* p = xyz + ((size_t)b * N_ + sidx[j]) * 3;
            float ctr = (c == 0) ? cx : ((c == 1) ? cy : cz);
            v = (p[c] - ctr) * (1.0f / RADIUS_);
        }
        h0x[j][c] = (_Float16)v;
    }

    // Wave 0: build gather-mode D# in SGPRs and issue the TDM load.
    if (nt == 0) {
        // Pack 16 x 16-bit row indices into 8 dwords (D# groups 2 and 3).
        unsigned p0, p1, p2, p3, p4, p5, p6, p7;
        {
            const int* ig = idx + g * NSAMPLE_;
            unsigned q[8];
#pragma unroll
            for (int j2 = 0; j2 < 8; ++j2) {
                unsigned lo = (unsigned)ig[2 * j2] & 0xFFFFu;
                unsigned hi = (unsigned)ig[2 * j2 + 1] & 0xFFFFu;
                q[j2] = __builtin_amdgcn_readfirstlane((int)(lo | (hi << 16)));
            }
            p0 = q[0]; p1 = q[1]; p2 = q[2]; p3 = q[3];
            p4 = q[4]; p5 = q[5]; p6 = q[6]; p7 = q[7];
        }
        unsigned lds_off =
            (unsigned)__builtin_amdgcn_readfirstlane(
                (int)(unsigned)(unsigned long long)(lds_f16_t*)&h0feat[0][0]);
        unsigned long long ga =
            (unsigned long long)(featT + (size_t)b * N_ * FD_);

        u32x4 d0;
        d0[0] = 0x80000001u;                       // count=1, gather, 16b idx
        d0[1] = lds_off;                           // LDS destination
        d0[2] = (unsigned)(ga & 0xFFFFFFFFu);      // global_addr[31:0]
        d0[3] = (unsigned)((ga >> 32) & 0x01FFFFFFu) | (2u << 30); // [56:32]|type=2

        u32x8 d1;
        d1[0] = (1u << 16);                        // data_size = 2 bytes
        d1[1] = ((unsigned)FD_ & 0xFFFFu) << 16;   // tensor_dim0 lo16 (256)
        d1[2] = ((unsigned)FD_ >> 16) | (((unsigned)N_ & 0xFFFFu) << 16); // dim0 hi | dim1 lo
        d1[3] = ((unsigned)N_ >> 16) | ((unsigned)FD_ << 16);  // dim1 hi | tile_dim0=256
        d1[4] = (unsigned)NSAMPLE_;                // tile_dim1 = 16 indices
        d1[5] = (unsigned)FD_;                     // tensor_dim0_stride lo32
        d1[6] = 0u;
        d1[7] = 0u;

        u32x4 d2, d3;
        d2[0] = p0; d2[1] = p1; d2[2] = p2; d2[3] = p3;
        d3[0] = p4; d3[1] = p5; d3[2] = p6; d3[3] = p7;

        asm volatile("tensor_load_to_lds %0, %1, %2, %3"
                     :: "s"(d0), "s"(d1), "s"(d2), "s"(d3)
                     : "memory");
        __builtin_amdgcn_s_wait_tensorcnt(0);
    }
    __syncthreads();   // h0feat (TDM) + h0x ready for all waves

    // ---- layer 1: [16x288] x [288x16] -> relu -> h1 ----
    v8f acc1 = {};
#pragma unroll
    for (int kt = 0; kt < 8; ++kt) {               // feature k-tiles (TDM tile)
        const int K0 = kt * 32 + half * 16;
        v8h lo = *(const v8h*)&h0feat[ncol][K0];
        v8h hi = *(const v8h*)&h0feat[ncol][K0 + 8];
        v16h a = __builtin_shufflevector(lo, hi, 0,1,2,3,4,5,6,7,8,9,10,11,12,13,14,15);
        v16h bb = *(const v16h*)&w1f[((nt * KT1 + kt) * 32 + lane) * 16];
        acc1 = __builtin_amdgcn_wmma_f32_16x16x32_f16(false, a, false, bb,
                                                      (short)0, acc1, false, false);
    }
    {                                              // xyz/pad k-tile (kt == 8)
        const int K0 = half * 16;
        v8h lo = *(const v8h*)&h0x[ncol][K0];
        v8h hi = *(const v8h*)&h0x[ncol][K0 + 8];
        v16h a = __builtin_shufflevector(lo, hi, 0,1,2,3,4,5,6,7,8,9,10,11,12,13,14,15);
        v16h bb = *(const v16h*)&w1f[((nt * KT1 + 8) * 32 + lane) * 16];
        acc1 = __builtin_amdgcn_wmma_f32_16x16x32_f16(false, a, false, bb,
                                                      (short)0, acc1, false, false);
    }
    {
        float bias = b1[nt * 16 + ncol];
#pragma unroll
        for (int r = 0; r < 8; ++r) {
            float v = acc1[r] + bias;
            v = v > 0.0f ? v : 0.0f;
            h1[half * 8 + r][nt * 16 + ncol] = (_Float16)v;
        }
    }
    __syncthreads();

    // ---- layer 2: [16x128] x [128x16] -> relu -> h2 ----
    v8f acc2 = {};
#pragma unroll
    for (int kt = 0; kt < KT2; ++kt) {
        const int K0 = kt * 32 + half * 16;
        v8h lo = *(const v8h*)&h1[ncol][K0];
        v8h hi = *(const v8h*)&h1[ncol][K0 + 8];
        v16h a = __builtin_shufflevector(lo, hi, 0,1,2,3,4,5,6,7,8,9,10,11,12,13,14,15);
        v16h bb = *(const v16h*)&w2f[((nt * KT2 + kt) * 32 + lane) * 16];
        acc2 = __builtin_amdgcn_wmma_f32_16x16x32_f16(false, a, false, bb,
                                                      (short)0, acc2, false, false);
    }
    {
        float bias = b2[nt * 16 + ncol];
#pragma unroll
        for (int r = 0; r < 8; ++r) {
            float v = acc2[r] + bias;
            v = v > 0.0f ? v : 0.0f;
            h2[half * 8 + r][nt * 16 + ncol] = (_Float16)v;
        }
    }
    __syncthreads();

    // ---- layer 3 + max over the 16 samples (rows) ----
    v8f acc3 = {};
#pragma unroll
    for (int kt = 0; kt < KT2; ++kt) {
        const int K0 = kt * 32 + half * 16;
        v8h lo = *(const v8h*)&h2[ncol][K0];
        v8h hi = *(const v8h*)&h2[ncol][K0 + 8];
        v16h a = __builtin_shufflevector(lo, hi, 0,1,2,3,4,5,6,7,8,9,10,11,12,13,14,15);
        v16h bb = *(const v16h*)&w3f[((nt * KT2 + kt) * 32 + lane) * 16];
        acc3 = __builtin_amdgcn_wmma_f32_16x16x32_f16(false, a, false, bb,
                                                      (short)0, acc3, false, false);
    }
    {
        float bias = b3[nt * 16 + ncol];
        float m = -1e30f;
#pragma unroll
        for (int r = 0; r < 8; ++r) {
            float v = acc3[r] + bias;
            v = v > 0.0f ? v : 0.0f;
            m = fmaxf(m, v);
        }
        float o = __shfl_xor(m, 16);     // combine M=0..7 half with M=8..15 half
        m = fmaxf(m, o);
        if (lane < 16) agg[(size_t)g * 128 + nt * 16 + ncol] = m;
    }
}

// ---------------------------------------------------------------------------
// Kernel 5: stable partition by mask (0s first / 1s first) + gather outputs.
// d_out flat layout (floats): vis_xyz | vis_features | vis_inds | mask_xyz | mask_inds
// ---------------------------------------------------------------------------
__launch_bounds__(256)
__global__ void select_kernel(const float* __restrict__ nxyz,
                              const float* __restrict__ agg,
                              const int* __restrict__ fp2,
                              const int* __restrict__ mask,
                              float* __restrict__ out) {
    __shared__ short vis[512], msk[512];
    const int b = blockIdx.x, t = threadIdx.x;
    const int K = NPOINT_ / 2;
    if (t == 0) {
        int nv = 0, nm = 0;
        for (int i = 0; i < NPOINT_; ++i) {
            if (mask[b * NPOINT_ + i]) { if (nm < K) msk[nm++] = (short)i; }
            else                       { if (nv < K) vis[nv++] = (short)i; }
        }
        while (nv < K) vis[nv++] = 0;
        while (nm < K) msk[nm++] = 0;
    }
    __syncthreads();
    const int OFF_VXYZ = 0;
    const int OFF_VF   = B_ * K * 3;                        // 12288
    const int OFF_VI   = OFF_VF + B_ * K * 128;             // 536576
    const int OFF_MX   = OFF_VI + B_ * K;                   // 540672
    const int OFF_MI   = OFF_MX + B_ * K * 3;               // 552960
    for (int k = t; k < K; k += 256) {
        int iv = vis[k], im = msk[k];
        for (int d = 0; d < 3; ++d) {
            out[OFF_VXYZ + (b * K + k) * 3 + d] = nxyz[(b * NPOINT_ + iv) * 3 + d];
            out[OFF_MX   + (b * K + k) * 3 + d] = nxyz[(b * NPOINT_ + im) * 3 + d];
        }
        out[OFF_VI + b * K + k] = (float)fp2[b * N_ + iv];
        out[OFF_MI + b * K + k] = (float)fp2[b * N_ + im];
    }
    for (int e = t; e < K * 128; e += 256) {
        int k = e >> 7, c = e & 127;
        out[OFF_VF + (b * K + k) * 128 + c] =
            agg[((size_t)b * NPOINT_ + vis[k]) * 128 + c];
    }
}

// ---------------------------------------------------------------------------
extern "C" void kernel_launch(void* const* d_in, const int* in_sizes, int n_in,
                              void* d_out, int out_size, void* d_ws, size_t ws_size,
                              hipStream_t stream) {
    const float* xyz  = (const float*)d_in[0];
    const float* feat = (const float*)d_in[1];
    const int*   fp2  = (const int*)d_in[2];
    const int*   mask = (const int*)d_in[3];
    const float* W1   = (const float*)d_in[4];
    const float* b1   = (const float*)d_in[5];
    const float* W2   = (const float*)d_in[6];
    const float* b2   = (const float*)d_in[7];
    const float* W3   = (const float*)d_in[8];
    const float* b3   = (const float*)d_in[9];

    char* ws = (char*)d_ws;
    int*      inds  = (int*)(ws + 0);                //   32768 B
    float*    nxyz  = (float*)(ws + 32768);          //   98304 B
    int*      idx   = (int*)(ws + 131072);           //  524288 B
    float*    agg   = (float*)(ws + 655360);         // 4194304 B
    _Float16* w1f   = (_Float16*)(ws + 4849664);     //   73728 B
    _Float16* w2f   = (_Float16*)(ws + 4923392);     //   32768 B
    _Float16* w3f   = (_Float16*)(ws + 4956160);     //   32768 B
    _Float16* featT = (_Float16*)(ws + 4988928);     // 8388608 B

    fps_kernel<<<B_, 256, 0, stream>>>(xyz, inds, nxyz);
    prep_feat_kernel<<<(B_ * N_ * FD_) / 256, 256, 0, stream>>>(feat, featT);
    prep_weights_kernel<<<(8 * KT1 * 32 * 16 + 2 * 8 * KT2 * 32 * 16 + 255) / 256,
                          256, 0, stream>>>(W1, W2, W3, w1f, w2f, w3f);
    ballquery_kernel<<<NGROUP_ * 32 / 256, 256, 0, stream>>>(xyz, nxyz, idx);
    mlp_kernel<<<NGROUP_, 256, 0, stream>>>(xyz, featT, nxyz, idx,
                                            w1f, w2f, w3f, b1, b2, b3, agg);
    select_kernel<<<B_, 256, 0, stream>>>(nxyz, agg, fp2, mask, (float*)d_out);
}
